// AttentionDecoder_57690000719931
// MI455X (gfx1250) — matrix operand order repaired and checked by reading, not compiled, not run on hardware
//
#include <hip/hip_runtime.h>
#include <hip/hip_bf16.h>

// Problem dims (from reference)
#define B_  32
#define T_  128
#define S_  512
#define M_  512
#define V_  10000

typedef __attribute__((ext_vector_type(16))) _Float16 v16h;
typedef __attribute__((ext_vector_type(8)))  float    v8f;

// ---------------------------------------------------------------------------
// WMMA fragment helpers (CDNA5 wave32 layouts, ISA 7.12.2)
// ---------------------------------------------------------------------------

// A tile: 16x32 (MxK), f16, row-major source with stride lda.
// lanes 0-15 -> M=lane, VGPR v holds K = (v<4 ? 2v : 8+2v); lanes 16-31: +8.
__device__ __forceinline__ v16h load_a_tile(const _Float16* A, int lda, int lane) {
  const int m = lane & 15;
  const int half = lane >> 4;
  v16h f;
#pragma unroll
  for (int v = 0; v < 8; ++v) {
    const int k = ((v < 4) ? (2 * v) : (8 + 2 * v)) + half * 8;
    f[2 * v]     = A[(size_t)m * lda + k];
    f[2 * v + 1] = A[(size_t)m * lda + k + 1];
  }
  return f;
}

// B tile: 32x16 (KxN) sourced as W^T where W is row-major [N,K] with stride ldw.
// Wp must already point at W + n0*ldw + k0.
// lanes 0-15 -> N=lane, VGPR v holds K = (v<4 ? 2v : 8+2v); lanes 16-31: +8.
__device__ __forceinline__ v16h load_bT_tile(const _Float16* Wp, int ldw, int lane) {
  const int n = lane & 15;
  const int half = lane >> 4;
  v16h f;
#pragma unroll
  for (int v = 0; v < 8; ++v) {
    const int k = ((v < 4) ? (2 * v) : (8 + 2 * v)) + half * 8;
    f[2 * v]     = Wp[(size_t)n * ldw + k];
    f[2 * v + 1] = Wp[(size_t)n * ldw + k + 1];
  }
  return f;
}

__device__ __forceinline__ v8f wmma16(v16h a, v16h b, v8f c) {
  return __builtin_amdgcn_wmma_f32_16x16x32_f16(false, a, false, b, (short)0, c,
                                                false, false);
}

__device__ __forceinline__ float sigmoidf_(float x) {
  return 1.0f / (1.0f + __expf(-x));
}

// ---------------------------------------------------------------------------
// Utility kernels
// ---------------------------------------------------------------------------

__global__ void cvt_f32_to_f16_kernel(_Float16* dst, const float* src, size_t n) {
  size_t i = (size_t)blockIdx.x * blockDim.x + threadIdx.x;
  size_t stride = (size_t)gridDim.x * blockDim.x;
  for (; i < n; i += stride) dst[i] = (_Float16)src[i];
}

// Embedding gather with padding_idx=0, one (b,t) row per block of 512 threads.
__global__ void embed_kernel(_Float16* emb, const int* tokens, const float* table) {
  const int row = blockIdx.x;           // b*T + t
  const int f = threadIdx.x;            // 0..511
  const int tok = tokens[row];
  const float v = (tok == 0) ? 0.0f : table[(size_t)tok * M_ + f];
  emb[(size_t)row * M_ + f] = (_Float16)v;
}

// Zero initial h (f32+f16, parity buffer 0) and ctx (f16).
__global__ void init_state_kernel(float* h_f32_0, _Float16* h_f16_0, _Float16* ctx_f16) {
  const int i = blockIdx.x * blockDim.x + threadIdx.x;   // 0 .. B_*M_-1
  if (i < B_ * M_) {
    h_f32_0[i] = 0.0f;
    h_f16_0[i] = (_Float16)0.0f;
    ctx_f16[i] = (_Float16)0.0f;
  }
}

// ---------------------------------------------------------------------------
// keys/values = features @ Wk^T + bk, features @ Wv^T + bv   (fused, WMMA)
// grid: (B_*S_/16, M_/16), block: 32 (one wave per 16x16 output tile)
// A fragment is reused across the K and V accumulator chains.
// ---------------------------------------------------------------------------
__global__ void kv_kernel(_Float16* __restrict__ keys, _Float16* __restrict__ values,
                          const _Float16* __restrict__ feat,
                          const _Float16* __restrict__ Wk, const _Float16* __restrict__ Wv,
                          const float* __restrict__ bk, const float* __restrict__ bv) {
  const int lane = threadIdx.x;
  const int row0 = blockIdx.x * 16;     // flattened (b,s)
  const int n0 = blockIdx.y * 16;
  v8f ak = {}, av = {};
  for (int kt = 0; kt < M_ / 32; ++kt) {
    v16h a = load_a_tile(feat + (size_t)row0 * M_ + kt * 32, M_, lane);
    ak = wmma16(a, load_bT_tile(Wk + (size_t)n0 * M_ + kt * 32, M_, lane), ak);
    av = wmma16(a, load_bT_tile(Wv + (size_t)n0 * M_ + kt * 32, M_, lane), av);
  }
  const int n = lane & 15, mb = (lane >> 4) * 8;
  const int col = n0 + n;
  const float bkc = bk[col], bvc = bv[col];
#pragma unroll
  for (int v = 0; v < 8; ++v) {
    const size_t idx = (size_t)(row0 + mb + v) * M_ + col;
    keys[idx] = (_Float16)(ak[v] + bkc);
    values[idx] = (_Float16)(av[v] + bvc);
  }
}

// ---------------------------------------------------------------------------
// Fused GRU cell step: gx = [emb_t, ctx] @ W_ih^T, gh = h @ W_hh^T, gates, h'.
// 6 WMMA accumulator chains per tile (r/z/n for input & hidden paths).
// grid: (B_/16, M_/16), block: 32
// ---------------------------------------------------------------------------
__global__ void gru_kernel(float* __restrict__ h_out, _Float16* __restrict__ h16_out,
                           _Float16* __restrict__ y16,
                           const float* __restrict__ h_in, const _Float16* __restrict__ h16_in,
                           const _Float16* __restrict__ ctx16,
                           const _Float16* __restrict__ emb,
                           const _Float16* __restrict__ Wih, const _Float16* __restrict__ Whh,
                           const float* __restrict__ b_ih, const float* __restrict__ b_hh,
                           int t) {
  const int lane = threadIdx.x;
  const int row0 = blockIdx.x * 16;     // batch rows
  const int n0 = blockIdx.y * 16;       // hidden-unit columns j
  v8f xr = {}, xz = {}, xn = {}, hr = {}, hz = {}, hn = {};

  // input path: K = 2*M_ = 1024 (first 512 from emb_t, second 512 from context)
  for (int kt = 0; kt < 2 * M_ / 32; ++kt) {
    const _Float16* ap;
    int lda;
    if (kt < M_ / 32) {
      ap = emb + (size_t)row0 * (T_ * M_) + (size_t)t * M_ + kt * 32;
      lda = T_ * M_;
    } else {
      ap = ctx16 + (size_t)row0 * M_ + (kt - M_ / 32) * 32;
      lda = M_;
    }
    v16h a = load_a_tile(ap, lda, lane);
    xr = wmma16(a, load_bT_tile(Wih + (size_t)(n0) * (2 * M_) + kt * 32, 2 * M_, lane), xr);
    xz = wmma16(a, load_bT_tile(Wih + (size_t)(M_ + n0) * (2 * M_) + kt * 32, 2 * M_, lane), xz);
    xn = wmma16(a, load_bT_tile(Wih + (size_t)(2 * M_ + n0) * (2 * M_) + kt * 32, 2 * M_, lane), xn);
  }
  // hidden path: K = 512
  for (int kt = 0; kt < M_ / 32; ++kt) {
    v16h a = load_a_tile(h16_in + (size_t)row0 * M_ + kt * 32, M_, lane);
    hr = wmma16(a, load_bT_tile(Whh + (size_t)(n0) * M_ + kt * 32, M_, lane), hr);
    hz = wmma16(a, load_bT_tile(Whh + (size_t)(M_ + n0) * M_ + kt * 32, M_, lane), hz);
    hn = wmma16(a, load_bT_tile(Whh + (size_t)(2 * M_ + n0) * M_ + kt * 32, M_, lane), hn);
  }

  const int n = lane & 15, mb = (lane >> 4) * 8;
  const int j = n0 + n;
  const float bir = b_ih[j], biz = b_ih[M_ + j], bin = b_ih[2 * M_ + j];
  const float bhr = b_hh[j], bhz = b_hh[M_ + j], bhn = b_hh[2 * M_ + j];
#pragma unroll
  for (int v = 0; v < 8; ++v) {
    const int gb = row0 + mb + v;       // batch index
    const float r = sigmoidf_(xr[v] + bir + hr[v] + bhr);
    const float z = sigmoidf_(xz[v] + biz + hz[v] + bhz);
    const float nn = tanhf(xn[v] + bin + r * (hn[v] + bhn));
    const float hold = h_in[(size_t)gb * M_ + j];
    const float hnew = (1.0f - z) * nn + z * hold;
    h_out[(size_t)gb * M_ + j] = hnew;
    h16_out[(size_t)gb * M_ + j] = (_Float16)hnew;
    y16[((size_t)gb * T_ + t) * (2 * M_) + j] = (_Float16)hnew;   // y[:, :M] = h
  }
}

// ---------------------------------------------------------------------------
// q = h' @ Wq^T + bq     grid: (B_/16, M_/16), block: 32
// ---------------------------------------------------------------------------
__global__ void q_kernel(float* __restrict__ q,
                         const _Float16* __restrict__ h16,
                         const _Float16* __restrict__ Wq, const float* __restrict__ bq) {
  const int lane = threadIdx.x;
  const int row0 = blockIdx.x * 16;
  const int n0 = blockIdx.y * 16;
  v8f acc = {};
  for (int kt = 0; kt < M_ / 32; ++kt) {
    v16h a = load_a_tile(h16 + (size_t)row0 * M_ + kt * 32, M_, lane);
    acc = wmma16(a, load_bT_tile(Wq + (size_t)n0 * M_ + kt * 32, M_, lane), acc);
  }
  const int n = lane & 15, mb = (lane >> 4) * 8;
  const int col = n0 + n;
  const float b = bq[col];
#pragma unroll
  for (int v = 0; v < 8; ++v)
    q[(size_t)(row0 + mb + v) * M_ + col] = acc[v] + b;
}

// ---------------------------------------------------------------------------
// scores + masked softmax over S; writes attention weights to d_out[B,S,T]
// and to workspace. grid: B_, block: 256
// ---------------------------------------------------------------------------
__global__ void score_softmax_kernel(float* __restrict__ w_ws, float* __restrict__ w_out,
                                     const _Float16* __restrict__ keys,
                                     const float* __restrict__ q,
                                     const unsigned char* __restrict__ mask, int t) {
  __shared__ float sc[S_];
  __shared__ float red[256];
  const int b = blockIdx.x, tid = threadIdx.x;
  const float scale = 0.04419417382415922f;   // 1/sqrt(512)
  for (int s = tid; s < S_; s += 256) {
    const _Float16* kp = keys + ((size_t)b * S_ + s) * M_;
    const float* qp = q + (size_t)b * M_;
    float acc = 0.0f;
    for (int m = 0; m < M_; ++m) acc += (float)kp[m] * qp[m];
    acc *= scale;
    if (!mask[b * S_ + s]) acc = -3.0e38f;
    sc[s] = acc;
  }
  __syncthreads();
  float lm = fmaxf(sc[tid], sc[tid + 256]);
  red[tid] = lm;
  __syncthreads();
  for (int o = 128; o > 0; o >>= 1) {
    if (tid < o) red[tid] = fmaxf(red[tid], red[tid + o]);
    __syncthreads();
  }
  const float mx = red[0];
  __syncthreads();
  float ls = 0.0f;
  for (int s = tid; s < S_; s += 256) {
    const float e = __expf(sc[s] - mx);
    sc[s] = e;
    ls += e;
  }
  red[tid] = ls;
  __syncthreads();
  for (int o = 128; o > 0; o >>= 1) {
    if (tid < o) red[tid] += red[tid + o];
    __syncthreads();
  }
  const float inv = 1.0f / red[0];
  for (int s = tid; s < S_; s += 256) {
    const float wv = sc[s] * inv;
    w_ws[(size_t)b * S_ + s] = wv;
    w_out[(size_t)b * S_ * T_ + (size_t)s * T_ + t] = wv;   // [B,S,T]
  }
}

// ---------------------------------------------------------------------------
// context = w @ values; writes ctx (f16 for next-step GEMM) and y[:, M:2M].
// grid: B_, block: 512 (one m per thread)
// ---------------------------------------------------------------------------
__global__ void context_kernel(_Float16* __restrict__ ctx16, _Float16* __restrict__ y16,
                               const float* __restrict__ w_ws,
                               const _Float16* __restrict__ values, int t) {
  const int b = blockIdx.x, m = threadIdx.x;
  float acc = 0.0f;
  for (int s = 0; s < S_; ++s)
    acc += w_ws[(size_t)b * S_ + s] * (float)values[((size_t)b * S_ + s) * M_ + m];
  ctx16[(size_t)b * M_ + m] = (_Float16)acc;
  y16[((size_t)b * T_ + t) * (2 * M_) + M_ + m] = (_Float16)acc;
}

// ---------------------------------------------------------------------------
// logits = y @ W_out^T + b_out  (dominant GEMM: 4096 x 10000 x 1024, WMMA)
// Register-blocked: each wave computes a 32x80 macro-tile (2 row-tiles x
// 5 col-tiles = 10 accumulators). Per k-step: 2 A-frag + 5 B-frag loads feed
// 10 WMMAs (0.7 fragment loads per WMMA instead of 2.0).
// grid: (B_*T_/32 = 128, V_/80 = 125), block: 32
// ---------------------------------------------------------------------------
__global__ void logits_kernel(float* __restrict__ out,
                              const _Float16* __restrict__ y16,
                              const _Float16* __restrict__ Wout,
                              const float* __restrict__ b_out) {
  const int lane = threadIdx.x;
  const int row0 = blockIdx.x * 32;
  const int n0 = blockIdx.y * 80;
  v8f acc0[5] = {{}, {}, {}, {}, {}};
  v8f acc1[5] = {{}, {}, {}, {}, {}};
  for (int kt = 0; kt < 2 * M_ / 32; ++kt) {
    const _Float16* ap = y16 + (size_t)row0 * (2 * M_) + kt * 32;
    // prefetch next k-step's A stream (lowers to global_prefetch_b8)
    __builtin_prefetch(ap + 32, 0, 3);
    v16h a0 = load_a_tile(ap, 2 * M_, lane);
    v16h a1 = load_a_tile(ap + (size_t)16 * (2 * M_), 2 * M_, lane);
#pragma unroll
    for (int nb = 0; nb < 5; ++nb) {
      const _Float16* bp = Wout + (size_t)(n0 + nb * 16) * (2 * M_) + kt * 32;
      __builtin_prefetch(bp + 32, 0, 3);
      v16h b = load_bT_tile(bp, 2 * M_, lane);
      acc0[nb] = wmma16(a0, b, acc0[nb]);
      acc1[nb] = wmma16(a1, b, acc1[nb]);
    }
  }
  const int n = lane & 15, mb = (lane >> 4) * 8;
#pragma unroll
  for (int nb = 0; nb < 5; ++nb) {
    const int col = n0 + nb * 16 + n;
    const float bo = b_out[col];
#pragma unroll
    for (int v = 0; v < 8; ++v) {
      out[(size_t)(row0 + mb + v) * V_ + col]      = acc0[nb][v] + bo;
      out[(size_t)(row0 + 16 + mb + v) * V_ + col] = acc1[nb][v] + bo;
    }
  }
}

// ---------------------------------------------------------------------------
// Host launcher
// ---------------------------------------------------------------------------
extern "C" void kernel_launch(void* const* d_in, const int* in_sizes, int n_in,
                              void* d_out, int out_size, void* d_ws, size_t ws_size,
                              hipStream_t stream) {
  (void)in_sizes; (void)n_in; (void)out_size; (void)ws_size;
  const int*   tokens  = (const int*)d_in[0];
  const float* feat    = (const float*)d_in[1];
  const unsigned char* mask = (const unsigned char*)d_in[2];
  const float* table   = (const float*)d_in[3];
  const float* W_ih    = (const float*)d_in[4];
  const float* W_hh    = (const float*)d_in[5];
  const float* b_ih    = (const float*)d_in[6];
  const float* b_hh    = (const float*)d_in[7];
  const float* Wq      = (const float*)d_in[8];
  const float* bq      = (const float*)d_in[9];
  const float* Wk      = (const float*)d_in[10];
  const float* bk      = (const float*)d_in[11];
  const float* Wv      = (const float*)d_in[12];
  const float* bv      = (const float*)d_in[13];
  const float* W_out   = (const float*)d_in[14];
  const float* b_out   = (const float*)d_in[15];

  float* d_logits  = (float*)d_out;                                // [B,T,V]
  float* d_weights = d_logits + (size_t)B_ * T_ * V_;              // [B,S,T]

  // ---- workspace carve-up (bytes) ----
  char* ws = (char*)d_ws;
  size_t off = 0;
  auto carve = [&](size_t bytes) { char* p = ws + off; off += (bytes + 255) & ~(size_t)255; return p; };
  _Float16* feat16  = (_Float16*)carve((size_t)B_ * S_ * M_ * 2);      // 16 MB
  _Float16* emb16   = (_Float16*)carve((size_t)B_ * T_ * M_ * 2);      // 4 MB
  _Float16* Wq16    = (_Float16*)carve((size_t)M_ * M_ * 2);
  _Float16* Wk16    = (_Float16*)carve((size_t)M_ * M_ * 2);
  _Float16* Wv16    = (_Float16*)carve((size_t)M_ * M_ * 2);
  _Float16* Wih16   = (_Float16*)carve((size_t)3 * M_ * 2 * M_ * 2);   // 3 MB
  _Float16* Whh16   = (_Float16*)carve((size_t)3 * M_ * M_ * 2);       // 1.5 MB
  _Float16* Wout16  = (_Float16*)carve((size_t)V_ * 2 * M_ * 2);       // 20 MB
  _Float16* keys16  = (_Float16*)carve((size_t)B_ * S_ * M_ * 2);      // 16 MB
  _Float16* vals16  = (_Float16*)carve((size_t)B_ * S_ * M_ * 2);      // 16 MB
  _Float16* y16     = (_Float16*)carve((size_t)B_ * T_ * 2 * M_ * 2);  // 8 MB
  float*    h_f32[2]; _Float16* h_f16[2];
  h_f32[0] = (float*)carve((size_t)B_ * M_ * 4);
  h_f32[1] = (float*)carve((size_t)B_ * M_ * 4);
  h_f16[0] = (_Float16*)carve((size_t)B_ * M_ * 2);
  h_f16[1] = (_Float16*)carve((size_t)B_ * M_ * 2);
  _Float16* ctx16   = (_Float16*)carve((size_t)B_ * M_ * 2);
  float*    q_f32   = (float*)carve((size_t)B_ * M_ * 4);
  float*    w_ws    = (float*)carve((size_t)B_ * S_ * 4);

  // ---- f32 -> f16 conversions ----
  auto cvt = [&](_Float16* dst, const float* src, size_t n) {
    int blocks = (int)((n + 255) / 256);
    if (blocks > 4096) blocks = 4096;
    cvt_f32_to_f16_kernel<<<blocks, 256, 0, stream>>>(dst, src, n);
  };
  cvt(feat16, feat, (size_t)B_ * S_ * M_);
  cvt(Wq16, Wq, (size_t)M_ * M_);
  cvt(Wk16, Wk, (size_t)M_ * M_);
  cvt(Wv16, Wv, (size_t)M_ * M_);
  cvt(Wih16, W_ih, (size_t)3 * M_ * 2 * M_);
  cvt(Whh16, W_hh, (size_t)3 * M_ * M_);
  cvt(Wout16, W_out, (size_t)V_ * 2 * M_);

  // embedding gather + state init
  embed_kernel<<<B_ * T_, M_, 0, stream>>>(emb16, tokens, table);
  init_state_kernel<<<(B_ * M_ + 255) / 256, 256, 0, stream>>>(h_f32[0], h_f16[0], ctx16);

  // keys/values projection (WMMA)
  kv_kernel<<<dim3(B_ * S_ / 16, M_ / 16), 32, 0, stream>>>(keys16, vals16, feat16,
                                                            Wk16, Wv16, bk, bv);

  // sequential decode loop
  for (int t = 0; t < T_; ++t) {
    const int pi = t & 1, po = (t + 1) & 1;
    gru_kernel<<<dim3(B_ / 16, M_ / 16), 32, 0, stream>>>(
        h_f32[po], h_f16[po], y16, h_f32[pi], h_f16[pi], ctx16, emb16,
        Wih16, Whh16, b_ih, b_hh, t);
    q_kernel<<<dim3(B_ / 16, M_ / 16), 32, 0, stream>>>(q_f32, h_f16[po], Wq16, bq);
    score_softmax_kernel<<<B_, 256, 0, stream>>>(w_ws, d_weights, keys16, q_f32, mask, t);
    context_kernel<<<B_, M_, 0, stream>>>(ctx16, y16, w_ws, vals16, t);
  }

  // output projection (dominant GEMM, WMMA, register-blocked 32x80)
  logits_kernel<<<dim3(B_ * T_ / 32, V_ / 80), 32, 0, stream>>>(d_logits, y16, Wout16, b_out);
}
